// Transformer_17987323035749
// MI455X (gfx1250) — matrix operand order repaired and checked by reading, not compiled, not run on hardware
//
#include <hip/hip_runtime.h>

#define DEPTH 8
#define DIMC 512
#define HEADS 8
#define HEAD_DIM 64
#define INNER 512
#define BATCH 8
#define SEQ 1024
#define ROWS (BATCH*SEQ)   /* 8192 */
#define ATT_SCALE 8.0f
#define LN_EPS 1e-5f
#define NORM_EPS 1e-12f

typedef __attribute__((ext_vector_type(16))) __bf16        v16bf;
typedef __attribute__((ext_vector_type(8)))  float         v8f;
typedef __attribute__((ext_vector_type(4)))  unsigned int  u32x4;
typedef __attribute__((ext_vector_type(4)))  unsigned int  v4u;
typedef __attribute__((ext_vector_type(4)))  int           v4i;
typedef __attribute__((ext_vector_type(8)))  int           v8i;

#if defined(__has_builtin)
#  if __has_builtin(__builtin_amdgcn_tensor_load_to_lds)
#    define USE_TDM 1
#  endif
#endif

union Frag16 {
    v16bf bf;
    u32x4 q[2];
    unsigned short us[16];
};

__device__ __forceinline__ unsigned short f32_to_bf16(float f) {
    unsigned int u = __float_as_uint(f);
    u += 0x7fffu + ((u >> 16) & 1u);   // round-to-nearest-even
    return (unsigned short)(u >> 16);
}

// ---------------------------------------------------------------------------
// Block-wide reduction of two values (sum, sumsq); blockDim.x == 256 (8 waves)
// ---------------------------------------------------------------------------
__device__ __forceinline__ void block_reduce2(float& a, float& b) {
    #pragma unroll
    for (int m = 16; m >= 1; m >>= 1) { a += __shfl_xor(a, m); b += __shfl_xor(b, m); }
    __shared__ float sa[8], sb[8];
    const int wave = threadIdx.x >> 5, lane = threadIdx.x & 31;
    if (lane == 0) { sa[wave] = a; sb[wave] = b; }
    __syncthreads();
    float ta = (lane < 8) ? sa[lane] : 0.f;
    float tb = (lane < 8) ? sb[lane] : 0.f;
    #pragma unroll
    for (int m = 4; m >= 1; m >>= 1) { ta += __shfl_xor(ta, m); tb += __shfl_xor(tb, m); }
    a = __shfl(ta, 0); b = __shfl(tb, 0);
}

// ---------------------------------------------------------------------------
// GEMM: C(f32, MxN) = A(bf16, MxK row-major) * B(bf16, KxN row-major)
// block = 256 threads (8 waves), block tile 128x64, per-wave 16x64, K-step 32.
// A tile staged by the Tensor Data Mover (TENSORcnt); B tile transposed by
// the waves into LDS so B-fragments are contiguous ds_load_b128 pairs.
// ---------------------------------------------------------------------------
__global__ __launch_bounds__(256) void gemm_bf16_k(
        const unsigned short* __restrict__ A, int lda,
        const unsigned short* __restrict__ B, int ldb,
        float* __restrict__ C, int ldc, int K) {
    __shared__ __align__(16) unsigned short lsA[128 * 32];  // [row][k]
    __shared__ __align__(16) unsigned short lsB[64 * 32];   // [n][k]  (transposed)

    const int tid  = threadIdx.x;
    const int wave = tid >> 5;
    const int lane = tid & 31;
    const int l15  = lane & 15;
    const int lh   = lane >> 4;

    const int rowBlock = blockIdx.y * 128;
    const int colBlock = blockIdx.x * 64;

    v8f acc[4] = {};

    const int bk = tid >> 3;          // 0..31 (k within tile)
    const int bn = (tid & 7) * 8;     // 0..56 (n within tile)
#if !defined(USE_TDM)
    const int ar = tid >> 1;          // 0..127
    const int ah = (tid & 1) * 16;    // 0 or 16
#endif

    for (int k0 = 0; k0 < K; k0 += 32) {
#if defined(USE_TDM)
        // --- A tile via Tensor Data Mover: 128 rows x 32 cols of bf16 ---
        if (wave == 0) {
            const unsigned long long ga =
                (unsigned long long)(const void*)(A + (size_t)rowBlock * lda + k0);
            const unsigned lds_off = (unsigned)(unsigned long long)(void*)&lsA[0];
            const unsigned td0 = (unsigned)K;                 // tensor dim0 (elems)
            const unsigned td1 = (unsigned)(gridDim.y * 128); // tensor dim1 (rows)
            v4u g0;
            g0.x = 1u;                                        // count=1, user mode
            g0.y = lds_off;                                   // lds_addr (bytes)
            g0.z = (unsigned)ga;                              // global_addr[31:0]
            g0.w = (unsigned)((ga >> 32) & 0x01FFFFFFull) | 0x80000000u; // [56:32] | type=2
            v8i g1;
            g1[0] = 0x00010000;                               // data_size = 2 bytes
            g1[1] = (int)((td0 & 0xFFFFu) << 16);             // tensor_dim0 lo
            g1[2] = (int)((td0 >> 16) | ((td1 & 0xFFFFu) << 16)); // dim0 hi | dim1 lo
            g1[3] = (int)((td1 >> 16) | (32u << 16));         // dim1 hi | tile_dim0=32
            g1[4] = 128;                                      // tile_dim1 = 128
            g1[5] = (int)(unsigned)lda;                       // tensor_dim0_stride
            g1[6] = 0;
            g1[7] = 0;
            const v4i z4 = {0, 0, 0, 0};                      // 2D: groups 2/3 unused
            const v8i z8 = {0, 0, 0, 0, 0, 0, 0, 0};
            __builtin_amdgcn_tensor_load_to_lds(g0, g1, z4, z4, z8, 0);
        }
#else
        const unsigned short* ag = A + (size_t)(rowBlock + ar) * lda + k0 + ah;
        *(u32x4*)&lsA[ar * 32 + ah]     = *(const u32x4*)ag;
        *(u32x4*)&lsA[ar * 32 + ah + 8] = *(const u32x4*)(ag + 8);
#endif
        // --- B tile 32x64 -> transposed [n][k] ---
        union { u32x4 q; unsigned short s[8]; } bt;
        bt.q = *(const u32x4*)(B + (size_t)(k0 + bk) * ldb + colBlock + bn);
        if (k0 + 32 < K)
            __builtin_prefetch((const void*)(B + (size_t)(k0 + 32 + bk) * ldb + colBlock + bn), 0, 3);
        #pragma unroll
        for (int j = 0; j < 8; ++j) lsB[(bn + j) * 32 + bk] = bt.s[j];
#if defined(USE_TDM)
        if (wave == 0) __builtin_amdgcn_s_wait_tensorcnt(0);
#endif
        __syncthreads();

        // A fragment: lane = row, K split {0..7,16..23}/{8..15,24..31}
        Frag16 a;
        const unsigned short* apt = &lsA[(wave * 16 + l15) * 32 + lh * 8];
        a.q[0] = *(const u32x4*)apt;
        a.q[1] = *(const u32x4*)(apt + 16);
        #pragma unroll
        for (int t = 0; t < 4; ++t) {
            // B fragment: lane = col, K split {0..7,16..23}/{8..15,24..31}
            Frag16 b;
            const unsigned short* bpt = &lsB[(t * 16 + l15) * 32 + lh * 8];
            b.q[0] = *(const u32x4*)bpt;
            b.q[1] = *(const u32x4*)(bpt + 16);
            acc[t] = __builtin_amdgcn_wmma_f32_16x16x32_bf16(
                         false, a.bf, false, b.bf, (short)0, acc[t], false, false);
        }
        __syncthreads();
    }

    // store: uniform block base + 32-bit per-lane offsets (saddr+voffset form)
    float* Cb = C + (size_t)rowBlock * ldc + colBlock;
    const int rof = (wave * 16 + lh * 8) * ldc + l15;
    #pragma unroll
    for (int t = 0; t < 4; ++t) {
        #pragma unroll
        for (int r = 0; r < 8; ++r)
            Cb[rof + r * ldc + t * 16] = acc[t][r];
    }
}

// ---------------------------------------------------------------------------
// Fused flash-style attention. Q/K/V bf16 in (b,h,s,64); O bf16 in (b,s,inner).
// grid = (SEQ/128, HEADS, BATCH); 256 threads; wave handles 16 query rows.
// ---------------------------------------------------------------------------
__global__ __launch_bounds__(256) void attn_k(
        const unsigned short* __restrict__ Qb,
        const unsigned short* __restrict__ Kb,
        const unsigned short* __restrict__ Vb,
        unsigned short* __restrict__ Ob) {
    __shared__ __align__(16) unsigned short ldsV[64 * 32];       // [d][key]
    __shared__ __align__(16) unsigned short ldsP[8 * 16 * 32];   // per-wave 16x32

    const int tid  = threadIdx.x;
    const int wave = tid >> 5;
    const int lane = tid & 31;
    const int l15  = lane & 15;
    const int lh   = lane >> 4;
    const int head = blockIdx.y;
    const int bz   = blockIdx.z;

    const size_t bh   = (size_t)(bz * HEADS + head) * SEQ;
    const int    qrow = blockIdx.x * 128 + wave * 16;

    // Q fragments (d = 0..31 and 32..63), loaded once
    Frag16 aq[2];
    {
        const unsigned short* qp = Qb + (bh + qrow + l15) * HEAD_DIM;
        #pragma unroll
        for (int f = 0; f < 2; ++f) {
            aq[f].q[0] = *(const u32x4*)(qp + f * 32 + lh * 8);
            aq[f].q[1] = *(const u32x4*)(qp + f * 32 + 16 + lh * 8);
        }
    }

    v8f oacc[4] = {};
    float rmax[8], rsum[8];
    #pragma unroll
    for (int r = 0; r < 8; ++r) { rmax[r] = -1e30f; rsum[r] = 0.f; }

    const int vk = tid >> 3;          // 0..31 key
    const int vn = (tid & 7) * 8;     // 0..56 d

    for (int kb = 0; kb < SEQ; kb += 32) {
        // stage V block (32 keys x 64 d) transposed -> ldsV[d][key]
        union { u32x4 q; unsigned short s[8]; } vt;
        vt.q = *(const u32x4*)(Vb + (bh + kb + vk) * HEAD_DIM + vn);
        #pragma unroll
        for (int j = 0; j < 8; ++j) ldsV[(vn + j) * 32 + vk] = vt.s[j];
        __syncthreads();

        // S = Q @ K^T (K rows in memory ARE the B-fragment columns)
        v8f sacc[2] = {};
        #pragma unroll
        for (int jj = 0; jj < 2; ++jj) {
            #pragma unroll
            for (int f = 0; f < 2; ++f) {
                Frag16 bkf;
                const unsigned short* kp =
                    Kb + (bh + kb + jj * 16 + l15) * HEAD_DIM + f * 32 + lh * 8;
                bkf.q[0] = *(const u32x4*)kp;
                bkf.q[1] = *(const u32x4*)(kp + 16);
                sacc[jj] = __builtin_amdgcn_wmma_f32_16x16x32_bf16(
                               false, aq[f].bf, false, bkf.bf, (short)0, sacc[jj], false, false);
            }
        }

        // online softmax; row m = lh*8 + r lives in VGPR r across a 16-lane half
        #pragma unroll
        for (int r = 0; r < 8; ++r) {
            float s0 = sacc[0][r] * ATT_SCALE;
            float s1 = sacc[1][r] * ATT_SCALE;
            float mx = fmaxf(s0, s1);
            #pragma unroll
            for (int m = 8; m >= 1; m >>= 1) mx = fmaxf(mx, __shfl_xor(mx, m));
            const float mn   = fmaxf(rmax[r], mx);
            const float corr = __expf(rmax[r] - mn);
            rmax[r] = mn;
            rsum[r] *= corr;
            #pragma unroll
            for (int t = 0; t < 4; ++t) oacc[t][r] *= corr;
            const float p0 = __expf(s0 - mn);
            const float p1 = __expf(s1 - mn);
            float ps = p0 + p1;
            #pragma unroll
            for (int m = 8; m >= 1; m >>= 1) ps += __shfl_xor(ps, m);
            rsum[r] += ps;
            unsigned short* pw = &ldsP[wave * 512 + (lh * 8 + r) * 32];
            pw[l15]      = f32_to_bf16(p0);
            pw[16 + l15] = f32_to_bf16(p1);
        }
        asm volatile("s_wait_dscnt 0" ::: "memory");  // per-wave LDS write->read

        // P as A-fragment from per-wave LDS
        Frag16 apf;
        const unsigned short* pr = &ldsP[wave * 512 + l15 * 32 + lh * 8];
        apf.q[0] = *(const u32x4*)pr;
        apf.q[1] = *(const u32x4*)(pr + 16);

        // O += P @ V
        #pragma unroll
        for (int t = 0; t < 4; ++t) {
            Frag16 bv;
            const unsigned short* vp = &ldsV[(t * 16 + l15) * 32 + lh * 8];
            bv.q[0] = *(const u32x4*)vp;
            bv.q[1] = *(const u32x4*)(vp + 16);
            oacc[t] = __builtin_amdgcn_wmma_f32_16x16x32_bf16(
                          false, apf.bf, false, bv.bf, (short)0, oacc[t], false, false);
        }
        __syncthreads();
    }

    const size_t orow0 = (size_t)bz * SEQ + qrow + lh * 8;
    #pragma unroll
    for (int r = 0; r < 8; ++r) {
        const float inv = 1.f / rsum[r];
        #pragma unroll
        for (int t = 0; t < 4; ++t)
            Ob[(orow0 + r) * INNER + head * HEAD_DIM + t * 16 + l15] =
                f32_to_bf16(oacc[t][r] * inv);
    }
}

// ---------------------------------------------------------------------------
// LayerNorm(row, width W) -> bf16
// ---------------------------------------------------------------------------
template <int W>
__global__ __launch_bounds__(256) void ln_to_bf16_k(
        const float* __restrict__ x, const float* __restrict__ g,
        const float* __restrict__ bb, unsigned short* __restrict__ out) {
    constexpr int P = W / 256;
    const int row = blockIdx.x, tid = threadIdx.x;
    const float* xr = x + (size_t)row * W;
    float v[P]; float s = 0.f, ss = 0.f;
    #pragma unroll
    for (int i = 0; i < P; ++i) { float t = xr[tid + i * 256]; v[i] = t; s += t; ss += t * t; }
    block_reduce2(s, ss);
    const float mu   = s * (1.0f / W);
    const float rstd = rsqrtf(ss * (1.0f / W) - mu * mu + LN_EPS);
    unsigned short* orow = out + (size_t)row * W;
    #pragma unroll
    for (int i = 0; i < P; ++i) {
        const int c = tid + i * 256;
        orow[c] = f32_to_bf16((v[i] - mu) * rstd * g[c] + bb[c]);
    }
}

// t[row] += LN(x[row]) * g + b     (width 512)
__global__ __launch_bounds__(256) void ln512_residual_k(
        const float* __restrict__ x, const float* __restrict__ g,
        const float* __restrict__ bb, float* __restrict__ t) {
    const int row = blockIdx.x, tid = threadIdx.x;
    const float* xr = x + (size_t)row * DIMC;
    const float v0 = xr[tid], v1 = xr[tid + 256];
    float s = v0 + v1, ss = v0 * v0 + v1 * v1;
    block_reduce2(s, ss);
    const float mu   = s * (1.0f / DIMC);
    const float rstd = rsqrtf(ss * (1.0f / DIMC) - mu * mu + LN_EPS);
    float* tr = t + (size_t)row * DIMC;
    tr[tid]       += (v0 - mu) * rstd * g[tid] + bb[tid];
    tr[tid + 256] += (v1 - mu) * rstd * g[tid + 256] + bb[tid + 256];
}

// GELU(exact) then LayerNorm (width 1024) -> bf16
__global__ __launch_bounds__(256) void gelu_ln1024_k(
        const float* __restrict__ x, const float* __restrict__ g,
        const float* __restrict__ bb, unsigned short* __restrict__ out) {
    const int row = blockIdx.x, tid = threadIdx.x;
    const float* xr = x + (size_t)row * 1024;
    float v[4]; float s = 0.f, ss = 0.f;
    #pragma unroll
    for (int i = 0; i < 4; ++i) {
        float t = xr[tid + i * 256];
        t = 0.5f * t * (1.0f + erff(t * 0.70710678118654752f));
        v[i] = t; s += t; ss += t * t;
    }
    block_reduce2(s, ss);
    const float mu   = s * (1.0f / 1024.0f);
    const float rstd = rsqrtf(ss * (1.0f / 1024.0f) - mu * mu + LN_EPS);
    unsigned short* orow = out + (size_t)row * 1024;
    #pragma unroll
    for (int i = 0; i < 4; ++i) {
        const int c = tid + i * 256;
        orow[c] = f32_to_bf16((v[i] - mu) * rstd * g[c] + bb[c]);
    }
}

// qk l2-norm + scale; repack q,k,v into (b,h,s,64) bf16. 32 threads per (b,s,h).
__global__ __launch_bounds__(32) void qknorm_k(
        const float* __restrict__ qf, const float* __restrict__ kvf,
        const float* __restrict__ qs, const float* __restrict__ ks,
        unsigned short* __restrict__ qb, unsigned short* __restrict__ kb,
        unsigned short* __restrict__ vb) {
    const int idx = blockIdx.x;
    const int h  = idx % HEADS;
    const int si = (idx / HEADS) % SEQ;
    const int bi = idx / (HEADS * SEQ);
    const int lane = threadIdx.x;

    const float* qr = qf + ((size_t)bi * SEQ + si) * INNER + h * HEAD_DIM;
    const float* kr = kvf + ((size_t)bi * SEQ + si) * (2 * INNER) + h * HEAD_DIM;
    const float* vr = kr + INNER;
    const float q0 = qr[lane], q1 = qr[lane + 32];
    const float k0 = kr[lane], k1 = kr[lane + 32];
    const float v0 = vr[lane], v1 = vr[lane + 32];
    float sq = q0 * q0 + q1 * q1, sk = k0 * k0 + k1 * k1;
    #pragma unroll
    for (int m = 16; m >= 1; m >>= 1) { sq += __shfl_xor(sq, m); sk += __shfl_xor(sk, m); }
    const float rq = 1.f / fmaxf(sqrtf(sq), NORM_EPS);
    const float rk = 1.f / fmaxf(sqrtf(sk), NORM_EPS);
    const size_t ob = ((size_t)(bi * HEADS + h) * SEQ + si) * HEAD_DIM;
    qb[ob + lane]      = f32_to_bf16(q0 * rq * qs[lane]);
    qb[ob + lane + 32] = f32_to_bf16(q1 * rq * qs[lane + 32]);
    kb[ob + lane]      = f32_to_bf16(k0 * rk * ks[lane]);
    kb[ob + lane + 32] = f32_to_bf16(k1 * rk * ks[lane + 32]);
    vb[ob + lane]      = f32_to_bf16(v0);
    vb[ob + lane + 32] = f32_to_bf16(v1);
}

__global__ __launch_bounds__(256) void f32_to_bf16_conv_k(
        const float* __restrict__ src, unsigned short* __restrict__ dst, int n) {
    const int i = blockIdx.x * 256 + threadIdx.x;
    if (i < n) dst[i] = f32_to_bf16(src[i]);
}

__global__ __launch_bounds__(256) void nchw_to_nsc_k(
        const float* __restrict__ x, float* __restrict__ t) {
    const int i  = blockIdx.x * 256 + threadIdx.x;   // (b, s, c), c fastest
    const int ci = i & (DIMC - 1);
    const int si = (i >> 9) & (SEQ - 1);
    const int bi = i >> 19;
    t[i] = x[((size_t)bi * DIMC + ci) * SEQ + si];
}

__global__ __launch_bounds__(256) void nsc_to_nchw_k(
        const float* __restrict__ t, float* __restrict__ o) {
    const int i  = blockIdx.x * 256 + threadIdx.x;   // (b, c, s), s fastest
    const int si = i & (SEQ - 1);
    const int ci = (i >> 10) & (DIMC - 1);
    const int bi = i >> 19;
    o[i] = t[((size_t)bi * SEQ + si) * DIMC + ci];
}

__global__ __launch_bounds__(256) void add_k(float* __restrict__ t, const float* __restrict__ a) {
    const int i = blockIdx.x * 256 + threadIdx.x;
    t[i] += a[i];
}

// ---------------------------------------------------------------------------
extern "C" void kernel_launch(void* const* d_in, const int* in_sizes, int n_in,
                              void* d_out, int out_size, void* d_ws, size_t ws_size,
                              hipStream_t stream) {
    const float* x        = (const float*)d_in[0];
    const float* Wq       = (const float*)d_in[1];
    const float* Wkv      = (const float*)d_in[2];
    const float* Wo       = (const float*)d_in[3];
    const float* qsc      = (const float*)d_in[4];
    const float* ksc      = (const float*)d_in[5];
    const float* ln_in_g  = (const float*)d_in[6];
    const float* ln_in_b  = (const float*)d_in[7];
    const float* ln_out_g = (const float*)d_in[8];
    const float* ln_out_b = (const float*)d_in[9];
    const float* ff1_g    = (const float*)d_in[10];
    const float* ff1_b    = (const float*)d_in[11];
    const float* W1       = (const float*)d_in[12];
    const float* ff2_g    = (const float*)d_in[13];
    const float* ff2_b    = (const float*)d_in[14];
    const float* W2       = (const float*)d_in[15];

    char* ws = (char*)d_ws;
    size_t off = 0;
    auto alloc = [&](size_t bytes) {
        char* p = ws + off;
        off += (bytes + 255) & ~(size_t)255;
        return (void*)p;
    };
    unsigned short* wq_b  = (unsigned short*)alloc((size_t)DEPTH * DIMC * INNER * 2);
    unsigned short* wkv_b = (unsigned short*)alloc((size_t)DEPTH * DIMC * 2 * INNER * 2);
    unsigned short* wo_b  = (unsigned short*)alloc((size_t)DEPTH * INNER * DIMC * 2);
    unsigned short* w1_b  = (unsigned short*)alloc((size_t)DEPTH * DIMC * 2 * DIMC * 2);
    unsigned short* w2_b  = (unsigned short*)alloc((size_t)DEPTH * 2 * DIMC * DIMC * 2);
    float*          t     = (float*)alloc((size_t)ROWS * DIMC * 4);
    unsigned short* xn    = (unsigned short*)alloc((size_t)ROWS * 2 * DIMC * 2);
    float*          qf    = (float*)alloc((size_t)ROWS * INNER * 4);
    float*          kvf   = (float*)alloc((size_t)ROWS * 2 * INNER * 4);  // reused as FFN hidden
    unsigned short* qb2   = (unsigned short*)alloc((size_t)ROWS * INNER * 2);
    unsigned short* kb2   = (unsigned short*)alloc((size_t)ROWS * INNER * 2);
    unsigned short* vb2   = (unsigned short*)alloc((size_t)ROWS * INNER * 2);
    unsigned short* ao    = (unsigned short*)alloc((size_t)ROWS * INNER * 2);
    float*          c1    = (float*)alloc((size_t)ROWS * DIMC * 4);
    (void)ws_size; (void)n_in; (void)in_sizes; (void)out_size;

    // one-time weight conversion to bf16 (stays resident in 192MB L2)
    f32_to_bf16_conv_k<<<DEPTH * DIMC * INNER / 256, 256, 0, stream>>>(Wq,  wq_b,  DEPTH * DIMC * INNER);
    f32_to_bf16_conv_k<<<DEPTH * DIMC * 2 * INNER / 256, 256, 0, stream>>>(Wkv, wkv_b, DEPTH * DIMC * 2 * INNER);
    f32_to_bf16_conv_k<<<DEPTH * INNER * DIMC / 256, 256, 0, stream>>>(Wo,  wo_b,  DEPTH * INNER * DIMC);
    f32_to_bf16_conv_k<<<DEPTH * DIMC * 2 * DIMC / 256, 256, 0, stream>>>(W1,  w1_b,  DEPTH * DIMC * 2 * DIMC);
    f32_to_bf16_conv_k<<<DEPTH * 2 * DIMC * DIMC / 256, 256, 0, stream>>>(W2,  w2_b,  DEPTH * 2 * DIMC * DIMC);

    nchw_to_nsc_k<<<ROWS * DIMC / 256, 256, 0, stream>>>(x, t);

    const dim3 gN512(INNER / 64, ROWS / 128);
    const dim3 gN1024(2 * INNER / 64, ROWS / 128);

    for (int i = 0; i < DEPTH; ++i) {
        const unsigned short* wq = wq_b  + (size_t)i * DIMC * INNER;
        const unsigned short* wk = wkv_b + (size_t)i * DIMC * 2 * INNER;
        const unsigned short* wo = wo_b  + (size_t)i * INNER * DIMC;
        const unsigned short* w1 = w1_b  + (size_t)i * DIMC * 2 * DIMC;
        const unsigned short* w2 = w2_b  + (size_t)i * 2 * DIMC * DIMC;

        // attention block
        ln_to_bf16_k<512><<<ROWS, 256, 0, stream>>>(t, ln_in_g + i * DIMC, ln_in_b + i * DIMC, xn);
        gemm_bf16_k<<<gN512,  256, 0, stream>>>(xn, DIMC, wq, INNER,     qf,  INNER,     DIMC);
        gemm_bf16_k<<<gN1024, 256, 0, stream>>>(xn, DIMC, wk, 2 * INNER, kvf, 2 * INNER, DIMC);
        qknorm_k<<<BATCH * SEQ * HEADS, 32, 0, stream>>>(qf, kvf, qsc + i * HEAD_DIM, ksc + i * HEAD_DIM,
                                                         qb2, kb2, vb2);
        attn_k<<<dim3(SEQ / 128, HEADS, BATCH), 256, 0, stream>>>(qb2, kb2, vb2, ao);
        gemm_bf16_k<<<gN512, 256, 0, stream>>>(ao, INNER, wo, DIMC, c1, DIMC, INNER);
        ln512_residual_k<<<ROWS, 256, 0, stream>>>(c1, ln_out_g + i * DIMC, ln_out_b + i * DIMC, t);

        // FFN block
        ln_to_bf16_k<512><<<ROWS, 256, 0, stream>>>(t, ff1_g + i * DIMC, ff1_b + i * DIMC, xn);
        gemm_bf16_k<<<gN1024, 256, 0, stream>>>(xn, DIMC, w1, 2 * DIMC, kvf, 2 * DIMC, DIMC);
        gelu_ln1024_k<<<ROWS, 256, 0, stream>>>(kvf, ff2_g + i * 2 * DIMC, ff2_b + i * 2 * DIMC, xn);
        gemm_bf16_k<<<gN512, 256, 0, stream>>>(xn, 2 * DIMC, w2, DIMC, c1, DIMC, 2 * DIMC);
        add_k<<<ROWS * DIMC / 256, 256, 0, stream>>>(t, c1);
    }

    nsc_to_nchw_k<<<ROWS * DIMC / 256, 256, 0, stream>>>(t, (float*)d_out);
}
